// LSTM_11450382811377
// MI455X (gfx1250) — compile-verified, matrix operand order checked by reading, not run on hardware
//
#include <hip/hip_runtime.h>

typedef __attribute__((ext_vector_type(16))) _Float16     v16h;
typedef __attribute__((ext_vector_type(4)))  _Float16     v4h;
typedef __attribute__((ext_vector_type(8)))  float        v8f;
typedef __attribute__((ext_vector_type(4)))  unsigned int u32x4;
typedef __attribute__((ext_vector_type(4)))  int          i32x4;
typedef __attribute__((ext_vector_type(8)))  int          i32x8;

static constexpr int kH  = 1024;        // hidden
static constexpr int kL  = 512;         // seq len
static constexpr int kN  = 32;          // batch
static constexpr int kG4 = 4 * kH;      // 4096 gate outputs
static constexpr int kR  = kN * kL;     // 16384 rows
static constexpr int kJB = 16;          // hidden-column strip per block
static constexpr int kBlocks = kH / kJB; // 64 persistent blocks

// ---------------------------------------------------------------------------
// 16x32 f16 tile (row-major, leading dim ld) -> WMMA A/B lane layout:
// lane l -> row (l&15), K elements {h8..h8+7, 16+h8..16+h8+7}, h8=(l>>4)*8.
// ---------------------------------------------------------------------------
__device__ __forceinline__ v16h load_tile16(const _Float16* __restrict__ p, int ld) {
    const int lane = threadIdx.x & 31;
    const int row  = lane & 15;
    const int h8   = (lane >> 4) << 3;
    const _Float16* q = p + (size_t)row * ld + h8;
    union { v16h h; u32x4 q4[2]; } u;
    u.q4[0] = *(const u32x4*)(q);
    u.q4[1] = *(const u32x4*)(q + 16);
    return u.h;
}

__device__ __forceinline__ v8f wmma16(v16h a, v16h b, v8f c) {
    return __builtin_amdgcn_wmma_f32_16x16x32_f16(false, a, false, b,
                                                  (short)0, c, false, false);
}

__device__ __forceinline__ float sigmoidf(float x) {
    return 1.0f / (1.0f + __expf(-x));
}

// ---------------------------------------------------------------------------
// TDM: async 2D tile load Global->LDS via Tensor Data Mover descriptor.
// rows x rowlen elements of 2-byte data, row stride `stride` elements.
// ---------------------------------------------------------------------------
#if __has_builtin(__builtin_amdgcn_tensor_load_to_lds)
#define HAVE_TDM 1
__device__ __forceinline__ void tdm_load_2d_f16(unsigned int lds_addr,
                                                const void* gptr,
                                                int rows, int rowlen, int stride) {
    const unsigned long long ga = (unsigned long long)gptr;
    u32x4 g0;
    g0[0] = 1u;                                     // count=1, user descriptor
    g0[1] = lds_addr;                               // LDS byte address
    g0[2] = (unsigned int)(ga & 0xFFFFFFFFu);       // global_addr[31:0]
    g0[3] = (unsigned int)((ga >> 32) & 0x01FFFFFFu) | (2u << 30);  // type=2
    i32x8 g1;
    g1[0] = (1 << 16);                              // data_size=1 (2 bytes)
    g1[1] = (int)((rowlen & 0xFFFF) << 16);         // tensor_dim0[15:0]
    g1[2] = (int)(((unsigned)rowlen >> 16) | ((unsigned)(rows & 0xFFFF) << 16));
    g1[3] = (int)(((unsigned)rows >> 16) | ((unsigned)(rowlen & 0xFFFF) << 16)); // tile_dim0
    g1[4] = rows & 0xFFFF;                          // tile_dim1 (tile_dim2=0)
    g1[5] = stride;                                 // tensor_dim0_stride[31:0]
    g1[6] = 0;
    g1[7] = 0;
    i32x4 g2 = {0, 0, 0, 0};
    i32x4 g3 = {0, 0, 0, 0};
#if __clang_major__ >= 23
    i32x8 g4 = {0, 0, 0, 0, 0, 0, 0, 0};
    __builtin_amdgcn_tensor_load_to_lds(g0, g1, g2, g3, g4, 0);
#else
    __builtin_amdgcn_tensor_load_to_lds(g0, g1, g2, g3, 0);
#endif
}
#else
#define HAVE_TDM 0
#endif

// ---------------------------------------------------------------------------
// x16[r, :] = f16(emb[tokens[r], :])
// ---------------------------------------------------------------------------
__global__ void embed_kernel(const int* __restrict__ tokens,
                             const float* __restrict__ emb,
                             _Float16* __restrict__ x16) {
    const int r   = blockIdx.x;
    const int tok = tokens[r];
    const float*  src = emb + (size_t)tok * kH;
    _Float16*     dst = x16 + (size_t)r * kH;
    const int base = threadIdx.x * 4;
    float4 v = *(const float4*)(src + base);
    v4h o = { (_Float16)v.x, (_Float16)v.y, (_Float16)v.z, (_Float16)v.w };
    *(v4h*)(dst + base) = o;
}

// ---------------------------------------------------------------------------
// Split W (D,4,H,2H) f32 into Wh / Wx, f16, row-major (layer,4096,1024).
// ---------------------------------------------------------------------------
__global__ void wcvt_kernel(const float* __restrict__ W,
                            _Float16* __restrict__ wh,
                            _Float16* __restrict__ wx) {
    const long total  = 2L * kG4 * (2 * kH);
    const long stride = (long)gridDim.x * blockDim.x;
    for (long idx = (long)blockIdx.x * blockDim.x + threadIdx.x; idx < total; idx += stride) {
        const long per_layer = (long)kG4 * (2 * kH);
        const long d   = idx / per_layer;
        const long rem = idx - d * per_layer;
        const long n   = rem >> 11;
        const long k   = rem & 2047;
        const float v  = W[idx];
        if (k < kH) wh[(d * kG4 + n) * kH + k]        = (_Float16)v;
        else        wx[(d * kG4 + n) * kH + (k - kH)] = (_Float16)v;
    }
}

__global__ void zero_state_kernel(float* __restrict__ c, _Float16* __restrict__ h,
                                  unsigned int* __restrict__ bar) {
    const int i = blockIdx.x * blockDim.x + threadIdx.x;
    if (i < kN * kH) { c[i] = 0.0f; h[i] = (_Float16)0.0f; }
    if (i == 0) *bar = 0u;
}

// ---------------------------------------------------------------------------
// Input projection: Y[r,n] = sum_k X[r,k]*Wx[n,k] + bias[n].
// 4 waves/block; wave computes 16(M) x 64(N).
// Unroll-by-2 ping-pong pipeline: no cross-iteration register copies.
// ---------------------------------------------------------------------------
__global__ void __launch_bounds__(128) xproj_kernel(
        const _Float16* __restrict__ X,    // (kR, kH)
        const _Float16* __restrict__ Wx,   // (kG4, kH)
        const float*    __restrict__ bias, // (kG4)
        float*          __restrict__ Y) {  // (kR, kG4)
    const int lane = threadIdx.x & 31;
    const int wave = threadIdx.x >> 5;
    const int m0 = blockIdx.x * 16;
    const int n0 = blockIdx.y * 256 + wave * 64;

    const _Float16* Xrow = X + (size_t)m0 * kH;

    const v8f zero8 = {0.f,0.f,0.f,0.f,0.f,0.f,0.f,0.f};
    v8f acc[4] = {zero8, zero8, zero8, zero8};

    v16h aA = load_tile16(Xrow, kH);
    v16h bA[4];
#pragma unroll
    for (int t = 0; t < 4; ++t)
        bA[t] = load_tile16(Wx + (size_t)(n0 + 16 * t) * kH, kH);

    for (int k0 = 0; k0 < kH - 64; k0 += 64) {
        v16h aB = load_tile16(Xrow + k0 + 32, kH);
        v16h bB[4];
#pragma unroll
        for (int t = 0; t < 4; ++t)
            bB[t] = load_tile16(Wx + (size_t)(n0 + 16 * t) * kH + k0 + 32, kH);
#pragma unroll
        for (int t = 0; t < 4; ++t)
            acc[t] = wmma16(aA, bA[t], acc[t]);

        aA = load_tile16(Xrow + k0 + 64, kH);
#pragma unroll
        for (int t = 0; t < 4; ++t)
            bA[t] = load_tile16(Wx + (size_t)(n0 + 16 * t) * kH + k0 + 64, kH);
#pragma unroll
        for (int t = 0; t < 4; ++t)
            acc[t] = wmma16(aB, bB[t], acc[t]);
    }
    {   // peeled tail: slices kH-64 (in A bufs) and kH-32
        v16h aB = load_tile16(Xrow + kH - 32, kH);
        v16h bB[4];
#pragma unroll
        for (int t = 0; t < 4; ++t)
            bB[t] = load_tile16(Wx + (size_t)(n0 + 16 * t) * kH + kH - 32, kH);
#pragma unroll
        for (int t = 0; t < 4; ++t)
            acc[t] = wmma16(aA, bA[t], acc[t]);
#pragma unroll
        for (int t = 0; t < 4; ++t)
            acc[t] = wmma16(aB, bB[t], acc[t]);
    }

    const int nn = lane & 15;
    const int mh = (lane >> 4) << 3;
#pragma unroll
    for (int t = 0; t < 4; ++t) {
        const int ncol = n0 + 16 * t + nn;
        const float bv = bias[ncol];
#pragma unroll
        for (int rr = 0; rr < 8; ++rr) {
            Y[(size_t)(m0 + mh + rr) * kG4 + ncol] = acc[t][rr] + bv;
        }
    }
}

// ---------------------------------------------------------------------------
// Persistent recurrent kernel: one launch runs all 512 timesteps of a layer.
// Grid = 64 blocks (hidden strips of 16), block = 4 waves (one gate each).
// Wh slab (4 gates x 16 rows x 1024 K, f16 = 128 KB) staged once into LDS by
// the TDM, then reused from LDS for every step.  Steps are separated by a
// monotonic global barrier (atomicAdd + acquire spin + device fences).
// ---------------------------------------------------------------------------
__global__ void __launch_bounds__(128) lstm_persistent_kernel(
        const _Float16* __restrict__ Wh,     // (kG4, kH) f16, this layer
        const float*    __restrict__ Xproj,  // (kR, kG4) f32, bias folded in
        float*          __restrict__ C,      // (kN, kH) f32 cell state
        _Float16*       __restrict__ hbase,  // 2 x (kN, kH) f16, ping-pong
        _Float16*       __restrict__ Xnext,  // (kR, kH) f16 or null
        float*          __restrict__ OutF32, // (kR, kH) f32 or null
        unsigned int*   __restrict__ bar) {
    __shared__ _Float16 whlds[4][kJB][kH];   // 128 KB weight slab
    __shared__ float    gbuf[4][kN][kJB + 1];

    const int lane = threadIdx.x & 31;
    const int gate = threadIdx.x >> 5;
    const int j0   = blockIdx.x * kJB;

    // ---- stage Wh slab into LDS once (Tensor Data Mover) ----
#if HAVE_TDM
    if (threadIdx.x < 32) {
#pragma unroll
        for (int g = 0; g < 4; ++g) {
            tdm_load_2d_f16(
                (unsigned int)(unsigned long long)&whlds[g][0][0],
                Wh + (size_t)(g * kH + j0) * kH,
                kJB, kH, kH);
        }
        __builtin_amdgcn_s_wait_tensorcnt(0);
    }
#else
    for (int g = 0; g < 4; ++g) {
        const _Float16* s = Wh + (size_t)(g * kH + j0) * kH;
        for (int idx = threadIdx.x; idx < kJB * kH / 8; idx += blockDim.x)
            ((u32x4*)&whlds[g][0][0])[idx] = ((const u32x4*)s)[idx];
    }
#endif
    __syncthreads();

    const int row = lane & 15;
    const int h8  = (lane >> 4) << 3;
    const v8f zero8 = {0.f,0.f,0.f,0.f,0.f,0.f,0.f,0.f};

    // B tile straight from the LDS slab (ds_load_b128 x2)
    auto lds_tile = [&](int k0) -> v16h {
        union { v16h h; u32x4 q4[2]; } u;
        u.q4[0] = *(const u32x4*)&whlds[gate][row][k0 + h8];
        u.q4[1] = *(const u32x4*)&whlds[gate][row][k0 + h8 + 16];
        return u.h;
    };

    for (int t = 0; t < kL; ++t) {
        const _Float16* Hin  = hbase + (size_t)(t & 1) * kN * kH;
        _Float16*       Hout = hbase + (size_t)((t + 1) & 1) * kN * kH;

        v8f acc0 = zero8, acc1 = zero8;

        // Unroll-by-2 ping-pong: A from global (L2-resident h), B from LDS.
        v16h aA0 = load_tile16(Hin, kH);
        v16h aA1 = load_tile16(Hin + 16 * kH, kH);
        v16h bA  = lds_tile(0);

        for (int k0 = 0; k0 < kH - 64; k0 += 64) {
            v16h aB0 = load_tile16(Hin + k0 + 32, kH);
            v16h aB1 = load_tile16(Hin + 16 * kH + k0 + 32, kH);
            v16h bB  = lds_tile(k0 + 32);
            acc0 = wmma16(aA0, bA, acc0);
            acc1 = wmma16(aA1, bA, acc1);

            aA0 = load_tile16(Hin + k0 + 64, kH);
            aA1 = load_tile16(Hin + 16 * kH + k0 + 64, kH);
            bA  = lds_tile(k0 + 64);
            acc0 = wmma16(aB0, bB, acc0);
            acc1 = wmma16(aB1, bB, acc1);
        }
        {   // peeled tail: slices kH-64 (in A bufs) and kH-32
            v16h aB0 = load_tile16(Hin + kH - 32, kH);
            v16h aB1 = load_tile16(Hin + 16 * kH + kH - 32, kH);
            v16h bB  = lds_tile(kH - 32);
            acc0 = wmma16(aA0, bA, acc0);
            acc1 = wmma16(aA1, bA, acc1);
            acc0 = wmma16(aB0, bB, acc0);
            acc1 = wmma16(aB1, bB, acc1);
        }

        // g = recurrent + precomputed input projection (bias already folded)
        const int nn = lane & 15;
        const int mh = (lane >> 4) << 3;
        const int nbase = gate * kH + j0;
#pragma unroll
        for (int rr = 0; rr < 8; ++rr) {
            const int m0r = mh + rr;
            gbuf[gate][m0r][nn] = acc0[rr] +
                Xproj[(size_t)(m0r * kL + t) * kG4 + nbase + nn];
            gbuf[gate][16 + m0r][nn] = acc1[rr] +
                Xproj[(size_t)((16 + m0r) * kL + t) * kG4 + nbase + nn];
        }
        __syncthreads();

        // fused gate nonlinearity + cell update: 512 elems / 128 threads
#pragma unroll
        for (int e = 0; e < 4; ++e) {
            const int lin = threadIdx.x + e * 128;
            const int m  = lin >> 4;
            const int jj = lin & 15;
            const int j  = j0 + jj;
            const float f    = sigmoidf(gbuf[0][m][jj]);
            const float ig   = sigmoidf(gbuf[1][m][jj]);
            const float o    = sigmoidf(gbuf[2][m][jj]);
            const float cand = gbuf[3][m][jj];
            const float c = f * C[m * kH + j] + ig * cand;
            C[m * kH + j] = c;
            const float h = o * tanhf(c);
            Hout[m * kH + j] = (_Float16)h;
            if (Xnext)  Xnext [(size_t)(m * kL + t) * kH + j] = (_Float16)h;
            if (OutF32) OutF32[(size_t)(m * kL + t) * kH + j] = h;
        }

        // grid-wide barrier before the next step consumes Hout
        if (t != kL - 1) {
            __syncthreads();
            if (threadIdx.x == 0) {
                __threadfence();                       // publish Hout
                atomicAdd(bar, 1u);
                const unsigned int target = (unsigned int)gridDim.x * (t + 1);
                while (__hip_atomic_load(bar, __ATOMIC_ACQUIRE,
                                         __HIP_MEMORY_SCOPE_AGENT) < target) {
                    __builtin_amdgcn_s_sleep(1);
                }
            }
            __syncthreads();
            __threadfence();                           // discard stale h in WGP$
        }
    }
}

// ---------------------------------------------------------------------------
extern "C" void kernel_launch(void* const* d_in, const int* in_sizes, int n_in,
                              void* d_out, int out_size, void* d_ws, size_t ws_size,
                              hipStream_t stream) {
    const int*   tokens = (const int*)  d_in[0];
    const float* emb    = (const float*)d_in[1];
    const float* W      = (const float*)d_in[2];
    const float* b      = (const float*)d_in[3];
    float* out = (float*)d_out;

    char* ws = (char*)d_ws;
    size_t off = 0;
    auto carve = [&](size_t bytes) -> void* {
        void* p = ws + off;
        off = (off + bytes + 255) & ~(size_t)255;
        return p;
    };
    float*        xproj = (float*)   carve((size_t)kR * kG4 * sizeof(float));     // 256 MB
    _Float16*     x16   = (_Float16*)carve((size_t)kR * kH * sizeof(_Float16));   //  32 MB
    _Float16*     x2    = (_Float16*)carve((size_t)kR * kH * sizeof(_Float16));   //  32 MB
    _Float16*     wh    = (_Float16*)carve((size_t)2 * kG4 * kH * sizeof(_Float16));
    _Float16*     wx    = (_Float16*)carve((size_t)2 * kG4 * kH * sizeof(_Float16));
    _Float16*     hbuf  = (_Float16*)carve((size_t)2 * kN * kH * sizeof(_Float16));
    float*        cbuf  = (float*)   carve((size_t)kN * kH * sizeof(float));
    unsigned int* bar   = (unsigned int*)carve(256);

    embed_kernel<<<kR, 256, 0, stream>>>(tokens, emb, x16);
    wcvt_kernel<<<4096, 256, 0, stream>>>(W, wh, wx);

    for (int layer = 0; layer < 2; ++layer) {
        const _Float16* xin = (layer == 0) ? x16 : x2;
        zero_state_kernel<<<(kN * kH + 255) / 256, 256, 0, stream>>>(cbuf, hbuf, bar);
        xproj_kernel<<<dim3(kR / 16, kG4 / 256), 128, 0, stream>>>(
            xin, wx + (size_t)layer * kG4 * kH, b + (size_t)layer * kG4, xproj);
        lstm_persistent_kernel<<<kBlocks, 128, 0, stream>>>(
            wh + (size_t)layer * kG4 * kH, xproj, cbuf, hbuf,
            (layer == 0) ? x2 : nullptr,
            (layer == 1) ? out : nullptr,
            bar);
    }
    (void)in_sizes; (void)n_in; (void)out_size; (void)ws_size;
}
